// GnnPolicy_11312943857830
// MI455X (gfx1250) — matrix operand order repaired and checked by reading, not compiled, not run on hardware
//
#include <hip/hip_runtime.h>
#include <hip/hip_bf16.h>

typedef __attribute__((ext_vector_type(16))) _Float16 v16h;
typedef __attribute__((ext_vector_type(8)))  _Float16 v8h;
typedef __attribute__((ext_vector_type(8)))  float    v8f;

#define NN  512
#define OBSD 32
#define HD  128
#define NL  3

__device__ inline v8f vzero8f() {
  v8f r;
#pragma unroll
  for (int i = 0; i < 8; ++i) r[i] = 0.0f;
  return r;
}

// ---------------------------------------------------------------------------
// Encoder MLP: z = mlp3(obs) ; one block per node, 128 threads
// ---------------------------------------------------------------------------
__global__ void enc_kernel(const float* __restrict__ obs,
                           const float* __restrict__ W1, const float* __restrict__ b1,
                           const float* __restrict__ W2, const float* __restrict__ b2,
                           const float* __restrict__ W3, const float* __restrict__ b3,
                           float* __restrict__ z) {
  __shared__ float xo[OBSD];
  __shared__ float h1[HD];
  __shared__ float h2[HD];
  const int i = blockIdx.x, m = threadIdx.x;
  if (m < OBSD) xo[m] = obs[i * OBSD + m];
  __syncthreads();
  float s = b1[m];
  const float* w = W1 + m * OBSD;
#pragma unroll 8
  for (int k = 0; k < OBSD; ++k) s += xo[k] * w[k];
  h1[m] = fmaxf(s, 0.0f);
  __syncthreads();
  s = b2[m];
  w = W2 + m * HD;
#pragma unroll 8
  for (int k = 0; k < HD; ++k) s += h1[k] * w[k];
  h2[m] = fmaxf(s, 0.0f);
  __syncthreads();
  s = b3[m];
  w = W3 + m * HD;
#pragma unroll 8
  for (int k = 0; k < HD; ++k) s += h2[k] * w[k];
  z[i * HD + m] = s;
}

// ---------------------------------------------------------------------------
// A = z @ Wi^T ; B = z @ Wj^T  (Wi = msg_W1[:, :H], Wj = msg_W1[:, H:2H])
// ---------------------------------------------------------------------------
__global__ void ab_kernel(const float* __restrict__ z,
                          const float* __restrict__ W1l, // [128 x 257] row-major
                          float* __restrict__ A, float* __restrict__ B) {
  __shared__ float zr[HD];
  const int i = blockIdx.x, m = threadIdx.x;
  zr[m] = z[i * HD + m];
  __syncthreads();
  const float* wr = W1l + m * (2 * HD + 1);
  float sa = 0.0f, sb = 0.0f;
#pragma unroll 8
  for (int k = 0; k < HD; ++k) { sa += zr[k] * wr[k]; sb += zr[k] * wr[HD + k]; }
  A[i * HD + m] = sa;
  B[i * HD + m] = sb;
}

// ---------------------------------------------------------------------------
// f32 -> f16 weight conversion (msg_W2 / msg_W3 staged once)
// ---------------------------------------------------------------------------
__global__ void cvt_f16_kernel(const float* __restrict__ in, _Float16* __restrict__ out, int n) {
  int idx = blockIdx.x * 256 + threadIdx.x;
  if (idx < n) out[idx] = (_Float16)in[idx];
}

__global__ void zero_kernel(float* __restrict__ p, int n) {
  int idx = blockIdx.x * 256 + threadIdx.x;
  if (idx < n) p[idx] = 0.0f;
}

// ---------------------------------------------------------------------------
// Fused pairwise message MLP + masked sum over j.
// grid = (32 i-tiles, 8 j-chunks), block = 256 threads (8 waves).
// Each wave handles one j at a time: 16 pairs (16 i x 1 j) = WMMA M-rows.
// GEMM1: relu(A_i+B_j+dist*wd+b1) @ W2^T  -> h2 (via f16 LDS bounce)
// GEMM2: h2 @ W3^T -> m, masked, accumulated in registers over j.
// ---------------------------------------------------------------------------
__launch_bounds__(256, 1)
__global__ void pair_kernel(const float* __restrict__ Afull,
                            const float* __restrict__ Bfull,
                            const float* __restrict__ pos,
                            const float* __restrict__ W1l,    // for wd column
                            const float* __restrict__ b1,
                            const _Float16* __restrict__ w2h, // [128x128] f16 row-major
                            const float* __restrict__ b2,
                            const _Float16* __restrict__ w3h, // [128x128] f16 row-major
                            const float* __restrict__ b3,
                            float* __restrict__ msum) {
  __shared__ float Ai[16 * HD];
  __shared__ float wd_s[HD], b1_s[HD], b2_s[HD], b3_s[HD];
  __shared__ float px[16], py[16];
  __shared__ float bj[8][HD];
  __shared__ _Float16 h2buf[8][16 * HD];
  __shared__ float ms[16 * HD];

  const int t = threadIdx.x;
  const int lane = t & 31;
  const int wv = t >> 5;
  const int iBase = blockIdx.x * 16;
  const int jChunk = blockIdx.y;

  for (int idx = t; idx < 16 * HD; idx += 256) {
    Ai[idx] = Afull[(iBase + (idx >> 7)) * HD + (idx & 127)];
    ms[idx] = 0.0f;
  }
  if (t < HD) {
    wd_s[t] = W1l[t * (2 * HD + 1) + 2 * HD];
    b1_s[t] = b1[t];
    b2_s[t] = b2[t];
    b3_s[t] = b3[t];
  }
  if (t < 16) { px[t] = pos[2 * (iBase + t)]; py[t] = pos[2 * (iBase + t) + 1]; }
  __syncthreads();

  const int mrow  = lane & 15;   // WMMA A row / B column index
  const int khalf = lane >> 4;   // lane-half selects K sub-block
  const float aix = px[mrow], aiy = py[mrow];

  v8f msacc[8];
#pragma unroll
  for (int nt = 0; nt < 8; ++nt) msacc[nt] = vzero8f();

  const int jEnd = (jChunk + 1) * 64;
  for (int j = jChunk * 64 + wv; j < jEnd; j += 8) {
    // stage B[j,:] into this wave's LDS slot
#pragma unroll
    for (int tt = 0; tt < 4; ++tt) bj[wv][lane * 4 + tt] = Bfull[j * HD + lane * 4 + tt];
    asm volatile("s_wait_dscnt 0" ::: "memory");

    const float pjx = pos[2 * j], pjy = pos[2 * j + 1];
    const float dx = aix - pjx, dy = aiy - pjy;
    const float dm = (iBase + mrow == j) ? 0.0f : sqrtf(dx * dx + dy * dy);

    // ---------------- GEMM1: h2 = relu(h1 @ W2^T + b2) -------------------
    v8f acc[8];
#pragma unroll
    for (int nt = 0; nt < 8; ++nt) acc[nt] = vzero8f();
#pragma unroll
    for (int kt = 0; kt < 4; ++kt) {
      v16h a;
#pragma unroll
      for (int e = 0; e < 16; ++e) {
        const int k = kt * 32 + (khalf << 4) + e;
        const float v = Ai[mrow * HD + k] + bj[wv][k] + dm * wd_s[k] + b1_s[k];
        a[e] = (_Float16)fmaxf(v, 0.0f);
      }
#pragma unroll
      for (int nt = 0; nt < 8; ++nt) {
        // B operand: lanes 0-15 K=kt*32..+15, lanes 16-31 K=+16..+31 (contiguous)
        const v16h bmat = *(const v16h*)(w2h + (nt * 16 + mrow) * HD + kt * 32 + khalf * 16);
        acc[nt] = __builtin_amdgcn_wmma_f32_16x16x32_f16(
            false, a, false, bmat, (short)0, acc[nt], false, false);
      }
    }
    // epilogue: bias+relu, bounce through per-wave LDS as f16 (C layout -> A layout)
#pragma unroll
    for (int nt = 0; nt < 8; ++nt) {
      const int c = nt * 16 + mrow;
      const float bb = b2_s[c];
#pragma unroll
      for (int v = 0; v < 8; ++v) {
        const int r = v + 8 * khalf;
        h2buf[wv][r * HD + c] = (_Float16)fmaxf(acc[nt][v] + bb, 0.0f);
      }
    }
    asm volatile("s_wait_dscnt 0" ::: "memory");

    // ---------------- GEMM2: m = h2 @ W3^T + b3 --------------------------
#pragma unroll
    for (int nt = 0; nt < 8; ++nt) acc[nt] = vzero8f();
#pragma unroll
    for (int kt = 0; kt < 4; ++kt) {
      v16h a;
      const v8h* hrow = (const v8h*)&h2buf[wv][mrow * HD + kt * 32 + khalf * 16];
      const v8h alo = hrow[0];
      const v8h ahi = hrow[1];  // +8 halves
#pragma unroll
      for (int e = 0; e < 8; ++e) { a[e] = alo[e]; a[e + 8] = ahi[e]; }
#pragma unroll
      for (int nt = 0; nt < 8; ++nt) {
        const v16h bmat = *(const v16h*)(w3h + (nt * 16 + mrow) * HD + kt * 32 + khalf * 16);
        acc[nt] = __builtin_amdgcn_wmma_f32_16x16x32_f16(
            false, a, false, bmat, (short)0, acc[nt], false, false);
      }
    }
    // epilogue: bias, diagonal mask, accumulate message sum in registers
#pragma unroll
    for (int nt = 0; nt < 8; ++nt) {
      const int c = nt * 16 + mrow;
      const float bb = b3_s[c];
#pragma unroll
      for (int v = 0; v < 8; ++v) {
        const int r = v + 8 * khalf;
        float val = acc[nt][v] + bb;
        if (iBase + r == j) val = 0.0f;
        msacc[nt][v] += val;
      }
    }
  }

  // reduce the 8 waves' partial sums into the block's LDS tile
#pragma unroll
  for (int nt = 0; nt < 8; ++nt) {
    const int c = nt * 16 + mrow;
#pragma unroll
    for (int v = 0; v < 8; ++v) {
      const int r = v + 8 * khalf;
      atomicAdd(&ms[r * HD + c], msacc[nt][v]);
    }
  }
  __syncthreads();
  // accumulate block tile into global msum (8 j-chunk blocks per i-tile)
  for (int idx = t; idx < 16 * HD; idx += 256) {
    atomicAdd(&msum[(iBase + (idx >> 7)) * HD + (idx & 127)], ms[idx]);
  }
}

// ---------------------------------------------------------------------------
// Update MLP: z' = mlp3(concat(z, msum)) ; one block per node, 128 threads
// ---------------------------------------------------------------------------
__global__ void upd_kernel(const float* __restrict__ z,
                           const float* __restrict__ msum,
                           const float* __restrict__ W1, const float* __restrict__ b1,
                           const float* __restrict__ W2, const float* __restrict__ b2,
                           const float* __restrict__ W3, const float* __restrict__ b3,
                           float* __restrict__ zout) {
  __shared__ float x[2 * HD];
  __shared__ float h1[HD];
  __shared__ float h2[HD];
  const int i = blockIdx.x, m = threadIdx.x;
  x[m] = z[i * HD + m];
  x[HD + m] = msum[i * HD + m];
  __syncthreads();
  float s = b1[m];
  const float* wr = W1 + m * (2 * HD);
#pragma unroll 8
  for (int k = 0; k < 2 * HD; ++k) s += x[k] * wr[k];
  h1[m] = fmaxf(s, 0.0f);
  __syncthreads();
  s = b2[m];
  wr = W2 + m * HD;
#pragma unroll 8
  for (int k = 0; k < HD; ++k) s += h1[k] * wr[k];
  h2[m] = fmaxf(s, 0.0f);
  __syncthreads();
  s = b3[m];
  wr = W3 + m * HD;
#pragma unroll 8
  for (int k = 0; k < HD; ++k) s += h2[k] * wr[k];
  zout[i * HD + m] = s;
}

// ---------------------------------------------------------------------------
extern "C" void kernel_launch(void* const* d_in, const int* in_sizes, int n_in,
                              void* d_out, int out_size, void* d_ws, size_t ws_size,
                              hipStream_t stream) {
  (void)in_sizes; (void)n_in; (void)out_size; (void)ws_size;
  const float* obs    = (const float*)d_in[0];
  const float* pos    = (const float*)d_in[1];
  const float* enc_W1 = (const float*)d_in[2];
  const float* enc_b1 = (const float*)d_in[3];
  const float* enc_W2 = (const float*)d_in[4];
  const float* enc_b2 = (const float*)d_in[5];
  const float* enc_W3 = (const float*)d_in[6];
  const float* enc_b3 = (const float*)d_in[7];
  const float* msg_W1 = (const float*)d_in[8];
  const float* msg_b1 = (const float*)d_in[9];
  const float* msg_W2 = (const float*)d_in[10];
  const float* msg_b2 = (const float*)d_in[11];
  const float* msg_W3 = (const float*)d_in[12];
  const float* msg_b3 = (const float*)d_in[13];
  const float* upd_W1 = (const float*)d_in[14];
  const float* upd_b1 = (const float*)d_in[15];
  const float* upd_W2 = (const float*)d_in[16];
  const float* upd_b2 = (const float*)d_in[17];
  const float* upd_W3 = (const float*)d_in[18];
  const float* upd_b3 = (const float*)d_in[19];

  float* ws   = (float*)d_ws;
  float* z0   = ws;                 // 512*128
  float* z1   = z0 + NN * HD;       // 512*128
  float* Abuf = z1 + NN * HD;       // 512*128
  float* Bbuf = Abuf + NN * HD;     // 512*128
  float* msum = Bbuf + NN * HD;     // 512*128
  _Float16* w2h = (_Float16*)(msum + NN * HD);   // 3*128*128 halves
  _Float16* w3h = w2h + NL * HD * HD;            // 3*128*128 halves

  const int nW = NL * HD * HD;
  cvt_f16_kernel<<<(nW + 255) / 256, 256, 0, stream>>>(msg_W2, w2h, nW);
  cvt_f16_kernel<<<(nW + 255) / 256, 256, 0, stream>>>(msg_W3, w3h, nW);

  enc_kernel<<<NN, HD, 0, stream>>>(obs, enc_W1, enc_b1, enc_W2, enc_b2, enc_W3, enc_b3, z0);

  float* zc = z0;
  for (int l = 0; l < NL; ++l) {
    const float* W1l = msg_W1 + (size_t)l * HD * (2 * HD + 1);
    ab_kernel<<<NN, HD, 0, stream>>>(zc, W1l, Abuf, Bbuf);
    zero_kernel<<<(NN * HD + 255) / 256, 256, 0, stream>>>(msum, NN * HD);
    dim3 pgrid(NN / 16, 8);
    pair_kernel<<<pgrid, 256, 0, stream>>>(
        Abuf, Bbuf, pos, W1l, msg_b1 + l * HD,
        w2h + (size_t)l * HD * HD, msg_b2 + l * HD,
        w3h + (size_t)l * HD * HD, msg_b3 + l * HD, msum);
    float* zn = (l == NL - 1) ? (float*)d_out : ((zc == z0) ? z1 : z0);
    upd_kernel<<<NN, HD, 0, stream>>>(
        zc, msum,
        upd_W1 + (size_t)l * HD * 2 * HD, upd_b1 + l * HD,
        upd_W2 + (size_t)l * HD * HD,     upd_b2 + l * HD,
        upd_W3 + (size_t)l * HD * HD,     upd_b3 + l * HD, zn);
    zc = zn;
  }
}
